// SOLD2_detector_23940147708556
// MI455X (gfx1250) — compile-verified, matrix operand order checked by reading, not run on hardware
//
#include <hip/hip_runtime.h>

// CDNA5 (gfx1250) SOLD2 line detector.
// One wave = 16 junction pairs; lane l computes 32 of the 64 segment samples
// for pair p = l%16, laid out per the 16-bit A-matrix 16x32 WMMA layout.
// Inlier counting runs on v_wmma_f32_16x16x32_f16 (exact: 0/1 indicators,
// integer f32 accumulation). Mean stays f32 with a ds_bpermute lane-pair add.
// Symmetry feat[j,i,s] == feat[i,j,63-s] lets us compute only j>=i and mirror,
// halving the (L2-bound) gather traffic.

typedef __attribute__((ext_vector_type(16))) _Float16 v16h;
typedef __attribute__((ext_vector_type(8)))  float    v8f;

#define NJ 1024
#define HH 1024
#define WWI 1024
#define DET_T 0.5f

__device__ __forceinline__ float bilin(const float* __restrict__ hm, float ch, float cw) {
    // Exactly replicates reference: floor/ceil corners, (hc-ch)/(ch-hf) weights.
    float hf = floorf(ch), hc = ceilf(ch);
    float wf = floorf(cw), wc = ceilf(cw);
    int hfi = (int)hf, hci = (int)hc, wfi = (int)wf, wci = (int)wc;
    float w00 = (hc - ch) * (wc - cw);
    float w01 = (hc - ch) * (cw - wf);
    float w10 = (ch - hf) * (wc - cw);
    float w11 = (ch - hf) * (cw - wf);
    return hm[hfi * WWI + wfi] * w00 + hm[hfi * WWI + wci] * w01 +
           hm[hci * WWI + wfi] * w10 + hm[hci * WWI + wci] * w11;
}

__global__ __launch_bounds__(256) void sold2_detect_kernel(
    const float* __restrict__ junc,   // [NJ,2] (h,w)
    const float* __restrict__ hm,     // [HH,WWI]
    float* __restrict__ out)          // [NJ*NJ line_map][NJ*NJ mean_feat]
{
    // Block tile: i in [by*16, by*16+16), j in [bx*8, bx*8+8).
    // Cull blocks strictly below the diagonal (uniform branch, EXEC stays full).
    if (blockIdx.x * 8 + 7 < blockIdx.y * 16) return;

    const int lane = threadIdx.x & 31;
    const int wave = threadIdx.x >> 5;          // 0..7 -> j within tile
    const int p    = lane & 15;                 // WMMA A/D row -> i within tile
    const int i    = blockIdx.y * 16 + p;
    const int j    = blockIdx.x * 8 + wave;

    const float hi_ = junc[2 * i];
    const float wi_ = junc[2 * i + 1];
    const float hj_ = junc[2 * j];
    const float wj_ = junc[2 * j + 1];

    // Lane l covers samples {sbase..sbase+15} (chunk0) and {32+sbase..} (chunk1);
    // lane pair (l, l+16) together covers all 64 samples of row p.
    const int sbase = (lane >> 4) << 4;         // 0 or 16

    float sum = 0.0f;
    v16h a0, a1;

#pragma unroll
    for (int e = 0; e < 16; ++e) {
        float t = (float)(sbase + e) / 63.0f;
        float u = 1.0f - t;
        float ch = fminf(fmaxf(hi_ * t + hj_ * u, 0.0f), (float)(HH - 1));
        float cw = fminf(fmaxf(wi_ * t + wj_ * u, 0.0f), (float)(WWI - 1));
        float f = bilin(hm, ch, cw);
        sum += f;
        a0[e] = (f > DET_T) ? (_Float16)1.0f : (_Float16)0.0f;
    }
#pragma unroll
    for (int e = 0; e < 16; ++e) {
        float t = (float)(32 + sbase + e) / 63.0f;
        float u = 1.0f - t;
        float ch = fminf(fmaxf(hi_ * t + hj_ * u, 0.0f), (float)(HH - 1));
        float cw = fminf(fmaxf(wi_ * t + wj_ * u, 0.0f), (float)(WWI - 1));
        float f = bilin(hm, ch, cw);
        sum += f;
        a1[e] = (f > DET_T) ? (_Float16)1.0f : (_Float16)0.0f;
    }

    // B = all ones: D[m][n] = sum_k A[m][k] -> exact inlier count per pair row.
    v16h bones;
#pragma unroll
    for (int e = 0; e < 16; ++e) bones[e] = (_Float16)1.0f;

    v8f d = {};
    d = __builtin_amdgcn_wmma_f32_16x16x32_f16(false, a0, false, bones,
                                               (short)0, d, false, false);
    d = __builtin_amdgcn_wmma_f32_16x16x32_f16(false, a1, false, bones,
                                               (short)0, d, false, false);

    // Lane-pair reduction of the f32 partial sums: lane l <-> lane l^16.
    int other_i = __builtin_amdgcn_ds_bpermute((lane ^ 16) << 2,
                                               __float_as_int(sum));
    float mean = (sum + __int_as_float(other_i)) * (1.0f / 64.0f);

    // D layout (f32 16x16): lanes 0-15 hold rows 0-7 in d[0..7],
    // lanes 16-31 hold rows 8-15. Writer lanes {0-7, 24-31} hold both the
    // count (d[lane&7]) and the mean of their pair p = lane%16.
    bool writer = (lane < 8) || (lane >= 24);
    if (writer && j >= i) {
        int r = lane & 7;
        float cnt = d[0];
        cnt = (r == 1) ? d[1] : cnt;
        cnt = (r == 2) ? d[2] : cnt;
        cnt = (r == 3) ? d[3] : cnt;
        cnt = (r == 4) ? d[4] : cnt;
        cnt = (r == 5) ? d[5] : cnt;
        cnt = (r == 6) ? d[6] : cnt;
        cnt = (r == 7) ? d[7] : cnt;

        bool det = (mean > DET_T) && (cnt * (1.0f / 64.0f) >= 0.99f) && (i < j);
        float lv = det ? 1.0f : 0.0f;

        out[i * NJ + j] = lv;                   // line_map (symmetric)
        out[j * NJ + i] = lv;
        float* om = out + NJ * NJ;              // mean_feat
        om[i * NJ + j] = mean;                  // mean is symmetric in (i,j)
        om[j * NJ + i] = mean;
    }
}

extern "C" void kernel_launch(void* const* d_in, const int* in_sizes, int n_in,
                              void* d_out, int out_size, void* d_ws, size_t ws_size,
                              hipStream_t stream) {
    (void)in_sizes; (void)n_in; (void)out_size; (void)d_ws; (void)ws_size;
    const float* junc = (const float*)d_in[0];   // [1024,2] f32
    const float* hm   = (const float*)d_in[1];   // [1024,1024] f32
    float* out        = (float*)d_out;           // 2 * 1024*1024 elements

    dim3 block(256, 1, 1);                       // 8 waves; wave = one j column
    dim3 grid(NJ / 8, NJ / 16, 1);               // x: j tiles (8), y: i tiles (16)
    sold2_detect_kernel<<<grid, block, 0, stream>>>(junc, hm, out);
}
